// MiniMaxExperts_74457553043834
// MI455X (gfx1250) — compile-verified, matrix operand order checked by reading, not run on hardware
//
#include <hip/hip_runtime.h>
#include <hip/hip_bf16.h>

// ---------------------------------------------------------------------------
// Grouped SwiGLU expert FFN for MI455X (gfx1250, wave32, WMMA).
// fp32 math emulated via split-bf16 (3x V_WMMA_F32_16X16X32_BF16 per tile).
// Software-pipelined: reg-prefetch next K-stage + double-buffered LDS +
// global_prefetch_b8 two stages ahead on the weight streams.
// ---------------------------------------------------------------------------

#define H_DIM 1024
#define I_DIM 2816
#define T_DIM 4096
#define E_NUM 8

#define BM 128          // tokens per block tile
#define BN 64           // output cols per block tile
#define BK 32           // K per stage == WMMA K for bf16
#define LDK 40          // padded LDS K stride in halves (80B, 16B aligned)

typedef __bf16 bf16_t;
typedef __attribute__((ext_vector_type(8)))  __bf16 v8bf;
typedef __attribute__((ext_vector_type(16))) __bf16 v16bf;
typedef __attribute__((ext_vector_type(8)))  float  v8f;

union Frag {
    v16bf v;
    v8bf  h[2];
};

__device__ __forceinline__ void split_bf16(float x, bf16_t& hi, bf16_t& lo) {
    hi = (bf16_t)x;                    // RNE
    lo = (bf16_t)(x - (float)hi);      // residual
}

__device__ __forceinline__ v8f wmma3(const Frag& ah, const Frag& al,
                                     const Frag& bh, const Frag& bl, v8f c) {
    // smallest terms first: alo*bhi + ahi*blo + ahi*bhi
    c = __builtin_amdgcn_wmma_f32_16x16x32_bf16(false, al.v, false, bh.v, (short)0, c, false, false);
    c = __builtin_amdgcn_wmma_f32_16x16x32_bf16(false, ah.v, false, bl.v, (short)0, c, false, false);
    c = __builtin_amdgcn_wmma_f32_16x16x32_bf16(false, ah.v, false, bh.v, (short)0, c, false, false);
    return c;
}

__device__ __forceinline__ void load_frag(Frag& f, const bf16_t* row, int khalf) {
    f.h[0] = *(const v8bf*)(row + khalf);
    f.h[1] = *(const v8bf*)(row + khalf + 16);
}

// ---------------------------------------------------------------------------
// Kernel 1: h = silu(x @ w1[e]) * (x @ w3[e]) for expert-owned token tile.
// Output written split into bf16 hi/lo planes (consumed by kernel 2).
// grid = (I/BN, T/BM, E), block = 256
// ---------------------------------------------------------------------------
__global__ __launch_bounds__(256)
void moe_up_gate_kernel(const float* __restrict__ x,
                        const int*   __restrict__ gsizes,
                        const float* __restrict__ w1,
                        const float* __restrict__ w3,
                        bf16_t* __restrict__ act_hi,
                        bf16_t* __restrict__ act_lo)
{
    const int e  = blockIdx.z;
    const int mt = blockIdx.y;
    const int nb = blockIdx.x * BN;

    int gStart = 0;
#pragma unroll
    for (int i = 0; i < E_NUM; ++i) gStart += (i < e) ? gsizes[i] : 0;
    const int ge = gsizes[e];
    if (mt * BM >= ge) return;                       // uniform per block
    const int rowsValid = min(BM, ge - mt * BM);
    const int tokBase   = gStart + mt * BM;

    const float* w1e = w1 + (size_t)e * H_DIM * I_DIM;
    const float* w3e = w3 + (size_t)e * H_DIM * I_DIM;

    __shared__ bf16_t Ah[2][BM][LDK];
    __shared__ bf16_t Al[2][BM][LDK];
    __shared__ bf16_t B1h[2][BN][LDK];
    __shared__ bf16_t B1l[2][BN][LDK];
    __shared__ bf16_t B3h[2][BN][LDK];
    __shared__ bf16_t B3l[2][BN][LDK];

    const int tid   = threadIdx.x;
    const int lane  = tid & 31;
    const int wave  = tid >> 5;          // 8 waves
    const int wm0   = (wave >> 1) * 32;  // 4 waves along M
    const int wn0   = (wave & 1)  * 32;  // 2 waves along N
    const int mrow  = lane & 15;
    const int khalf = (lane >> 4) * 8;   // 0 or 8

    v8f acc1[2][2] = {};
    v8f acc3[2][2] = {};

    float4 pa[4];            // reg-prefetch buffers (next K stage)
    float4 pb1[2], pb3[2];

    auto loadStage = [&](int k0) {
#pragma unroll
        for (int it = 0; it < 4; ++it) {
            int idx = (tid + it * 256) * 4;
            int r = idx >> 5, c = idx & 31;
            pa[it] = (r < rowsValid)
                   ? *(const float4*)(x + (size_t)(tokBase + r) * H_DIM + k0 + c)
                   : make_float4(0.f, 0.f, 0.f, 0.f);
        }
#pragma unroll
        for (int it = 0; it < 2; ++it) {
            int idx = (tid + it * 256) * 4;
            int kk = idx >> 6, nn = idx & 63;
            pb1[it] = *(const float4*)(w1e + (size_t)(k0 + kk) * I_DIM + nb + nn);
            pb3[it] = *(const float4*)(w3e + (size_t)(k0 + kk) * I_DIM + nb + nn);
        }
    };

    auto storeStage = [&](int buf) {
#pragma unroll
        for (int it = 0; it < 4; ++it) {
            int idx = (tid + it * 256) * 4;
            int r = idx >> 5, c = idx & 31;
#pragma unroll
            for (int j = 0; j < 4; ++j) {
                bf16_t hi, lo;
                split_bf16((&pa[it].x)[j], hi, lo);
                Ah[buf][r][c + j] = hi;
                Al[buf][r][c + j] = lo;
            }
        }
#pragma unroll
        for (int it = 0; it < 2; ++it) {
            int idx = (tid + it * 256) * 4;
            int kk = idx >> 6, nn = idx & 63;
#pragma unroll
            for (int j = 0; j < 4; ++j) {
                bf16_t hi, lo;
                split_bf16((&pb1[it].x)[j], hi, lo);
                B1h[buf][nn + j][kk] = hi;  B1l[buf][nn + j][kk] = lo;
                split_bf16((&pb3[it].x)[j], hi, lo);
                B3h[buf][nn + j][kk] = hi;  B3l[buf][nn + j][kk] = lo;
            }
        }
    };

    const int S = H_DIM / BK;           // 32 stages
    loadStage(0);
    storeStage(0);
    __syncthreads();

    for (int s = 0; s < S; ++s) {
        const int  buf  = s & 1;
        const bool more = (s + 1) < S;

        // issue next stage's global loads early (overlap with WMMA below)
        if (more) loadStage((s + 1) * BK);

        // L2 prefetch two stages ahead on the weight streams
        if (s + 2 < S) {
            const int k2 = (s + 2) * BK;
            const int kk = tid >> 3;            // 0..31
            const int nn = (tid & 7) * 8;       // 0..56
            __builtin_prefetch(w1e + (size_t)(k2 + kk) * I_DIM + nb + nn, 0, 3);
            __builtin_prefetch(w3e + (size_t)(k2 + kk) * I_DIM + nb + nn, 0, 3);
        }

        // ---- WMMA on current LDS buffer: 2x2 tiles x (w1,w3) x 3 terms ----
        Frag ah[2], al[2];
#pragma unroll
        for (int mi = 0; mi < 2; ++mi) {
            int r = wm0 + mi * 16 + mrow;
            load_frag(ah[mi], &Ah[buf][r][0], khalf);
            load_frag(al[mi], &Al[buf][r][0], khalf);
        }
#pragma unroll
        for (int ni = 0; ni < 2; ++ni) {
            int n = wn0 + ni * 16 + mrow;
            Frag bh, bl;
            load_frag(bh, &B1h[buf][n][0], khalf);
            load_frag(bl, &B1l[buf][n][0], khalf);
#pragma unroll
            for (int mi = 0; mi < 2; ++mi)
                acc1[mi][ni] = wmma3(ah[mi], al[mi], bh, bl, acc1[mi][ni]);
            load_frag(bh, &B3h[buf][n][0], khalf);
            load_frag(bl, &B3l[buf][n][0], khalf);
#pragma unroll
            for (int mi = 0; mi < 2; ++mi)
                acc3[mi][ni] = wmma3(ah[mi], al[mi], bh, bl, acc3[mi][ni]);
        }

        // stash prefetched tile into the other LDS buffer; one barrier/stage
        if (more) storeStage(buf ^ 1);
        __syncthreads();
    }

    // ---- fused SwiGLU epilogue, split-bf16 store of intermediate ----
    const int nOut  = lane & 15;
    const int mBase = (lane >> 4) * 8;
#pragma unroll
    for (int mi = 0; mi < 2; ++mi)
#pragma unroll
        for (int ni = 0; ni < 2; ++ni)
#pragma unroll
            for (int j = 0; j < 8; ++j) {
                int r = wm0 + mi * 16 + mBase + j;
                if (r < rowsValid) {
                    float f1 = acc1[mi][ni][j];
                    float f3 = acc3[mi][ni][j];
                    float hv = (f1 / (1.0f + __expf(-f1))) * f3;   // silu*gate
                    bf16_t hi, lo;
                    split_bf16(hv, hi, lo);
                    size_t off = (size_t)(tokBase + r) * I_DIM + nb + wn0 + ni * 16 + nOut;
                    act_hi[off] = hi;
                    act_lo[off] = lo;
                }
            }
}

// ---------------------------------------------------------------------------
// Kernel 2: out = h @ w2[e].  h already split bf16 hi/lo in d_ws.
// grid = (H/BN, T/BM, E), block = 256
// ---------------------------------------------------------------------------
__global__ __launch_bounds__(256)
void moe_down_kernel(const bf16_t* __restrict__ act_hi,
                     const bf16_t* __restrict__ act_lo,
                     const int*    __restrict__ gsizes,
                     const float*  __restrict__ w2,
                     float* __restrict__ out)
{
    const int e  = blockIdx.z;
    const int mt = blockIdx.y;
    const int nb = blockIdx.x * BN;

    int gStart = 0;
#pragma unroll
    for (int i = 0; i < E_NUM; ++i) gStart += (i < e) ? gsizes[i] : 0;
    const int ge = gsizes[e];
    if (mt * BM >= ge) return;
    const int rowsValid = min(BM, ge - mt * BM);
    const int tokBase   = gStart + mt * BM;

    const float* w2e = w2 + (size_t)e * I_DIM * H_DIM;

    __shared__ bf16_t Ah[2][BM][LDK];
    __shared__ bf16_t Al[2][BM][LDK];
    __shared__ bf16_t Bh[2][BN][LDK];
    __shared__ bf16_t Bl[2][BN][LDK];

    const int tid   = threadIdx.x;
    const int lane  = tid & 31;
    const int wave  = tid >> 5;
    const int wm0   = (wave >> 1) * 32;
    const int wn0   = (wave & 1)  * 32;
    const int mrow  = lane & 15;
    const int khalf = (lane >> 4) * 8;

    v8f acc[2][2] = {};

    uint4  pah[2], pal[2];   // reg-prefetch: pre-split bf16 activations
    float4 pb[2];            // reg-prefetch: fp32 w2 tile

    auto loadStage = [&](int k0) {
#pragma unroll
        for (int it = 0; it < 2; ++it) {
            int idx = (tid + it * 256) * 8;          // 8 halves per thread
            int r = idx >> 5, c = idx & 31;
            if (r < rowsValid) {
                size_t off = (size_t)(tokBase + r) * I_DIM + k0 + c;
                pah[it] = *(const uint4*)(act_hi + off);
                pal[it] = *(const uint4*)(act_lo + off);
            } else {
                pah[it] = make_uint4(0u, 0u, 0u, 0u);
                pal[it] = make_uint4(0u, 0u, 0u, 0u);
            }
        }
#pragma unroll
        for (int it = 0; it < 2; ++it) {
            int idx = (tid + it * 256) * 4;
            int kk = idx >> 6, nn = idx & 63;
            pb[it] = *(const float4*)(w2e + (size_t)(k0 + kk) * H_DIM + nb + nn);
        }
    };

    auto storeStage = [&](int buf) {
#pragma unroll
        for (int it = 0; it < 2; ++it) {
            int idx = (tid + it * 256) * 8;
            int r = idx >> 5, c = idx & 31;
            *(uint4*)&Ah[buf][r][c] = pah[it];
            *(uint4*)&Al[buf][r][c] = pal[it];
        }
#pragma unroll
        for (int it = 0; it < 2; ++it) {
            int idx = (tid + it * 256) * 4;
            int kk = idx >> 6, nn = idx & 63;
#pragma unroll
            for (int j = 0; j < 4; ++j) {
                bf16_t hi, lo;
                split_bf16((&pb[it].x)[j], hi, lo);
                Bh[buf][nn + j][kk] = hi;
                Bl[buf][nn + j][kk] = lo;
            }
        }
    };

    const int S = I_DIM / BK;            // 88 stages
    loadStage(0);
    storeStage(0);
    __syncthreads();

    for (int s = 0; s < S; ++s) {
        const int  buf  = s & 1;
        const bool more = (s + 1) < S;

        if (more) loadStage((s + 1) * BK);

        if (s + 2 < S) {
            const int k2 = (s + 2) * BK;
            const int kk = tid >> 3;
            const int nn = (tid & 7) * 8;
            __builtin_prefetch(w2e + (size_t)(k2 + kk) * H_DIM + nb + nn, 0, 3);
        }

        Frag ah[2], al[2];
#pragma unroll
        for (int mi = 0; mi < 2; ++mi) {
            int r = wm0 + mi * 16 + mrow;
            load_frag(ah[mi], &Ah[buf][r][0], khalf);
            load_frag(al[mi], &Al[buf][r][0], khalf);
        }
#pragma unroll
        for (int ni = 0; ni < 2; ++ni) {
            int n = wn0 + ni * 16 + mrow;
            Frag bh, bl;
            load_frag(bh, &Bh[buf][n][0], khalf);
            load_frag(bl, &Bl[buf][n][0], khalf);
#pragma unroll
            for (int mi = 0; mi < 2; ++mi)
                acc[mi][ni] = wmma3(ah[mi], al[mi], bh, bl, acc[mi][ni]);
        }

        if (more) storeStage(buf ^ 1);
        __syncthreads();
    }

    // ---- fp32 output ----
    const int nOut  = lane & 15;
    const int mBase = (lane >> 4) * 8;
#pragma unroll
    for (int mi = 0; mi < 2; ++mi)
#pragma unroll
        for (int ni = 0; ni < 2; ++ni)
#pragma unroll
            for (int j = 0; j < 8; ++j) {
                int r = wm0 + mi * 16 + mBase + j;
                if (r < rowsValid)
                    out[(size_t)(tokBase + r) * H_DIM + nb + wn0 + ni * 16 + nOut] =
                        acc[mi][ni][j];
            }
}

// ---------------------------------------------------------------------------
// Launch
// ---------------------------------------------------------------------------
extern "C" void kernel_launch(void* const* d_in, const int* in_sizes, int n_in,
                              void* d_out, int out_size, void* d_ws, size_t ws_size,
                              hipStream_t stream) {
    const float* x  = (const float*)d_in[0];   // [T,H]
    const int*   gs = (const int*)  d_in[1];   // [E]
    const float* w1 = (const float*)d_in[2];   // [E,H,I]
    const float* w2 = (const float*)d_in[3];   // [E,I,H]
    const float* w3 = (const float*)d_in[4];   // [E,H,I]
    float* out = (float*)d_out;                // [T,H]

    bf16_t* act_hi = (bf16_t*)d_ws;                              // [T,I] bf16
    bf16_t* act_lo = act_hi + (size_t)T_DIM * I_DIM;             // [T,I] bf16

    dim3 blk(256);
    dim3 g1(I_DIM / BN, T_DIM / BM, E_NUM);   // 44 x 32 x 8
    moe_up_gate_kernel<<<g1, blk, 0, stream>>>(x, gs, w1, w3, act_hi, act_lo);

    dim3 g2(H_DIM / BN, T_DIM / BM, E_NUM);   // 16 x 32 x 8
    moe_down_kernel<<<g2, blk, 0, stream>>>(act_hi, act_lo, gs, w2, out);
}
